// CCL_Loss_33165737460211
// MI455X (gfx1250) — matrix operand order repaired and checked by reading, not compiled, not run on hardware
//
#include <hip/hip_runtime.h>
#include <hip/hip_bf16.h>

typedef __attribute__((ext_vector_type(16))) _Float16 v16h;
typedef __attribute__((ext_vector_type(8)))  _Float16 v8h;
typedef __attribute__((ext_vector_type(8)))  float    v8f;

#define BB 512
#define VV 2
#define DD 128
#define MM 1024          // V*B
#define TOPK 15
#define KPAD 16          // pad 15 -> 16 so one j == one 16-wide WMMA tile
#define NROWS (BB * KPAD) // 8192
#define KMAXC 50
#define INV_T (1.0f / 0.07f)

// Fast 1-ULP hardware ops: v_sqrt_f32 / v_rcp_f32 (no IEEE refinement branches).
__device__ __forceinline__ float adc_from_sq(float sq) {
    float d = __builtin_amdgcn_sqrtf(fmaxf(sq, 0.0f));   // sqrt(0) == 0
    return (1.0f + __builtin_amdgcn_rcpf(1.0f + d)) * INV_T;
}

// ---------------- WMMA fragment loaders (row-major packed f16, ld = DD) ----
// A 16x32 f16 layout: lane L holds row M = L%16, h = L/16;
//   element e<8  -> K = 8h + e ; e>=8 -> K = 16 + 8h + (e-8)
__device__ __forceinline__ v16h load_a_frag(const _Float16* __restrict__ base,
                                            int row, int kbase, int lane) {
    int m = lane & 15, h = lane >> 4;
    const _Float16* p = base + (row + m) * DD + kbase;
    v8h lo = *(const v8h*)(p + 8 * h);
    v8h hi = *(const v8h*)(p + 16 + 8 * h);
    v16h r;
#pragma unroll
    for (int i = 0; i < 8; ++i) { r[i] = lo[i]; r[i + 8] = hi[i]; }
    return r;
}

// B 32x16 f16 layout: lane L holds col N = L%16, h = L/16;
//   element e<8  -> K = 8h + e ; e>=8 -> K = 16 + 8h + (e-8)
__device__ __forceinline__ v16h load_b_frag(const _Float16* __restrict__ base,
                                            int col, int kbase, int lane) {
    int n = lane & 15, h = lane >> 4;
    const _Float16* p = base + (col + n) * DD + kbase;
    v8h lo = *(const v8h*)(p + 8 * h);
    v8h hi = *(const v8h*)(p + 16 + 8 * h);
    v16h r;
#pragma unroll
    for (int i = 0; i < 8; ++i) { r[i] = lo[i]; r[i + 8] = hi[i]; }
    return r;
}

// ---------------- Kernel 1: anchor pack (swapaxes layout) + exact f32 norms --
__global__ void pack_anchor(const float* __restrict__ features,
                            _Float16* __restrict__ Ah, float* __restrict__ sqA) {
    int m = blockIdx.x;      // 0..1023 : row m = v*B + b  -> features[b, v, :]
    int d = threadIdx.x;     // 0..127
    int v = m / BB, b = m % BB;
    float x = features[(b * VV + v) * DD + d];
    Ah[m * DD + d] = (_Float16)x;
    __shared__ float red[DD];
    red[d] = x * x;
    __syncthreads();
    for (int s = DD / 2; s > 0; s >>= 1) {
        if (d < s) red[d] += red[d + s];
        __syncthreads();
    }
    if (d == 0) sqA[m] = red[0];
}

// ---------------- Kernel 2: neighbor gather/pack (k padded to 16) -----------
__global__ void pack_neigh(const float* __restrict__ saved,
                           const int* __restrict__ indices,
                           const int* __restrict__ rks,
                           _Float16* __restrict__ Nh, float* __restrict__ sqN) {
    int c = blockIdx.x;      // 0..8191 : c = j*16 + k
    int d = threadIdx.x;
    int j = c >> 4, k = c & 15;
    float x = 0.0f;
    if (k < TOPK) {
        long long row = (long long)rks[(long long)indices[j] * KMAXC + k];
        x = saved[row * DD + d];
    }
    Nh[c * DD + d] = (_Float16)x;
    __shared__ float red[DD];
    red[d] = x * x;
    __syncthreads();
    for (int s = DD / 2; s > 0; s >>= 1) {
        if (d < s) red[d] += red[d + s];
        __syncthreads();
    }
    if (d == 0) sqN[c] = red[0];
}

// ---------------- Kernel 3: G0 = A A^T via WMMA, fused adc_orig epilogue ----
__global__ void gemm0_adc0(const _Float16* __restrict__ Ah,
                           const float* __restrict__ sqA,
                           float* __restrict__ adc0) {
    int wave = threadIdx.x >> 5;
    int lane = threadIdx.x & 31;
    int tile = blockIdx.x * 8 + wave;   // 0..4095 tiles (64 x 64)
    int ti = tile >> 6, tj = tile & 63;
    v8f c = {};
#pragma unroll
    for (int kb = 0; kb < DD; kb += 32) {
        v16h a = load_a_frag(Ah, ti * 16, kb, lane);
        v16h b = load_b_frag(Ah, tj * 16, kb, lane);
        c = __builtin_amdgcn_wmma_f32_16x16x32_f16(false, a, false, b,
                                                   (short)0, c, false, false);
    }
    int n = lane & 15, h = lane >> 4;
    int j = tj * 16 + n;
    float sj = sqA[j];
#pragma unroll
    for (int r = 0; r < 8; ++r) {
        int i = ti * 16 + r + 8 * h;
        adc0[i * MM + j] = adc_from_sq(sqA[i] + sj - 2.0f * c[r]);
    }
}

// ---- Kernel 4: Gk = A N^T via WMMA, fused f(dist) + in-tile k-reduction ----
__global__ void gemmk_acc(const _Float16* __restrict__ Ah,
                          const float* __restrict__ sqA,
                          const _Float16* __restrict__ Nh,
                          const float* __restrict__ sqN,
                          float* __restrict__ acc) {
    int wave = threadIdx.x >> 5;
    int lane = threadIdx.x & 31;
    int tile = blockIdx.x * 8 + wave;   // 0..32767 tiles (64 i-tiles x 512 j)
    int ti = tile >> 9, tj = tile & 511;
    v8f c = {};
#pragma unroll
    for (int kb = 0; kb < DD; kb += 32) {
        v16h a = load_a_frag(Ah, ti * 16, kb, lane);
        v16h b = load_b_frag(Nh, tj * 16, kb, lane);
        c = __builtin_amdgcn_wmma_f32_16x16x32_f16(false, a, false, b,
                                                   (short)0, c, false, false);
    }
    int k = lane & 15, h = lane >> 4;
    float sn = sqN[tj * 16 + k];
#pragma unroll
    for (int r = 0; r < 8; ++r) {
        int i = ti * 16 + r + 8 * h;
        float f = (k < TOPK) ? adc_from_sq(sqA[i] + sn - 2.0f * c[r]) : 0.0f;
        // sum over the 16 k-columns: lanes with equal M sit in a 16-lane half
        f += __shfl_xor(f, 1, 32);
        f += __shfl_xor(f, 2, 32);
        f += __shfl_xor(f, 4, 32);
        f += __shfl_xor(f, 8, 32);
        if (k == 0) acc[i * BB + tj] = f * (1.0f / (float)TOPK);
    }
}

// ---------------- Kernel 5: per-row streaming logsumexp + positive pair -----
__global__ void row_loss(const float* __restrict__ acc,
                         const float* __restrict__ adc0,
                         float* __restrict__ loss) {
    int i = blockIdx.x;
    int t = threadIdx.x;
    int ib = i & (BB - 1);
    float m = -3.0e38f, s = 0.0f;
    for (int j = t; j < MM; j += 256) {
        float a1 = acc[i * BB + (j & (BB - 1))];
        float a2 = acc[j * BB + ib];
        float a0 = adc0[i * MM + j];
        float x = __builtin_amdgcn_sqrtf(a1 * a1 + a2 * a2 + a0 * a0);
        float nm = fmaxf(m, x);                 // max includes diagonal
        s = s * __expf(m - nm) + ((j != i) ? __expf(x - nm) : 0.0f); // sum excludes it
        m = nm;
    }
    __shared__ float sm[256], ss[256];
    sm[t] = m; ss[t] = s;
    __syncthreads();
    for (int st = 128; st > 0; st >>= 1) {
        if (t < st) {
            float m2 = sm[t + st], s2 = ss[t + st];
            float nm = fmaxf(sm[t], m2);
            ss[t] = ss[t] * __expf(sm[t] - nm) + s2 * __expf(m2 - nm);
            sm[t] = nm;
        }
        __syncthreads();
    }
    if (t == 0) {
        int jp = (i + BB) & (MM - 1);           // the one positive pair (V=2)
        float a1 = acc[i * BB + (jp & (BB - 1))];
        float a2 = acc[jp * BB + ib];
        float a0 = adc0[i * MM + jp];
        float xp = __builtin_amdgcn_sqrtf(a1 * a1 + a2 * a2 + a0 * a0);
        loss[i] = -(xp - sm[0] - __logf(ss[0]));
    }
}

// ---------------- Kernel 6: mean over 1024 rows -----------------------------
__global__ void final_mean(const float* __restrict__ loss, float* __restrict__ out) {
    __shared__ float red[256];
    int t = threadIdx.x;
    red[t] = loss[t] + loss[t + 256] + loss[t + 512] + loss[t + 768];
    __syncthreads();
    for (int st = 128; st > 0; st >>= 1) {
        if (t < st) red[t] += red[t + st];
        __syncthreads();
    }
    if (t == 0) out[0] = red[0] / (float)MM;
}

extern "C" void kernel_launch(void* const* d_in, const int* in_sizes, int n_in,
                              void* d_out, int out_size, void* d_ws, size_t ws_size,
                              hipStream_t stream) {
    const float* features = (const float*)d_in[0];   // (512, 2, 128) f32
    const int*   indices  = (const int*)d_in[1];     // (512,) i32
    const float* saved    = (const float*)d_in[2];   // (100000, 128) f32
    const int*   rks      = (const int*)d_in[3];     // (100000, 50) i32
    float* out = (float*)d_out;

    char* w = (char*)d_ws;
    _Float16* Ah  = (_Float16*)w; w += (size_t)MM * DD * sizeof(_Float16);     // 256 KB
    _Float16* Nh  = (_Float16*)w; w += (size_t)NROWS * DD * sizeof(_Float16);  // 2 MB
    float* sqA  = (float*)w; w += (size_t)MM * sizeof(float);
    float* sqN  = (float*)w; w += (size_t)NROWS * sizeof(float);
    float* adc0 = (float*)w; w += (size_t)MM * MM * sizeof(float);             // 4 MB
    float* accb = (float*)w; w += (size_t)MM * BB * sizeof(float);             // 2 MB
    float* loss = (float*)w; w += (size_t)MM * sizeof(float);

    pack_anchor<<<MM, DD, 0, stream>>>(features, Ah, sqA);
    pack_neigh<<<NROWS, DD, 0, stream>>>(saved, indices, rks, Nh, sqN);
    gemm0_adc0<<<(64 * 64) / 8, 256, 0, stream>>>(Ah, sqA, adc0);
    gemmk_acc<<<(64 * 512) / 8, 256, 0, stream>>>(Ah, sqA, Nh, sqN, accb);
    row_loss<<<MM, 256, 0, stream>>>(accb, adc0, loss);
    final_mean<<<1, 256, 0, stream>>>(loss, out);
}
